// GATLayer_68375879352698
// MI455X (gfx1250) — matrix operand order repaired and checked by reading, not compiled, not run on hardware
//
#include <hip/hip_runtime.h>
#include <hip/hip_bf16.h>

#define B_DIM 8
#define N_DIM 1024
#define H_DIM 512
#define NEG_SLOPE 0.2f
#define PSTR 1032  // padded LDS row stride (bf16 elems) for the probs tile
#define XSTR 520   // padded LDS row stride for the prep transpose staging

#if __has_builtin(__builtin_amdgcn_sched_barrier)
#define SCHED_FENCE() __builtin_amdgcn_sched_barrier(0)
#else
#define SCHED_FENCE()
#endif

typedef __attribute__((ext_vector_type(16))) __bf16 v16bf;
typedef __attribute__((ext_vector_type(8)))  float  v8f;

union V16 { uint4 q[2]; v16bf v; };
union P8 { unsigned short s[8]; uint4 q; };

__device__ __forceinline__ unsigned short f32_to_bf16(float f) {
    union { float f; unsigned u; } cv; cv.f = f;
    unsigned u = cv.u;
    u += 0x7FFFu + ((u >> 16) & 1u);   // round-to-nearest-even
    return (unsigned short)(u >> 16);
}

// ---------------------------------------------------------------------------
// Kernel 1: per-row dots s_src/s_dst + bf16 transpose of features Xt[b][h][j].
// 8 rows per 256-thread block; transpose staged through LDS so global stores
// are 16B-packed (8 consecutive j per h) instead of scattered 2B stores.
// ---------------------------------------------------------------------------
__global__ __launch_bounds__(256) void gat_prep(
    const float* __restrict__ x, const float* __restrict__ w_src,
    const float* __restrict__ w_dst, float* __restrict__ s_src,
    float* __restrict__ s_dst, unsigned short* __restrict__ xt)
{
    __shared__ unsigned short xs[8 * XSTR];

    const int wave = threadIdx.x >> 5;
    const int lane = threadIdx.x & 31;
    const int row  = blockIdx.x * 8 + wave;     // 0..8191  (b*N + n)
    const int bb   = row >> 10;
    const int n0   = (blockIdx.x * 8) & (N_DIM - 1);
    const float* xr = x + (size_t)row * H_DIM;

    float ss = 0.f, sd = 0.f;
#pragma unroll
    for (int k = 0; k < H_DIM / 32; ++k) {
        int h = lane + 32 * k;                  // coalesced across the wave
        float f = xr[h];
        ss += f * w_src[h];
        sd += f * w_dst[h];
        xs[wave * XSTR + h] = f32_to_bf16(f);
    }
#pragma unroll
    for (int off = 16; off > 0; off >>= 1) {
        ss += __shfl_down(ss, off, 32);
        sd += __shfl_down(sd, off, 32);
    }
    if (lane == 0) { s_src[row] = ss; s_dst[row] = sd; }
    __syncthreads();

    // LDS transpose: each thread packs 8 consecutive n for 2 h values.
#pragma unroll
    for (int rep = 0; rep < 2; ++rep) {
        int h = threadIdx.x + 256 * rep;
        P8 p;
#pragma unroll
        for (int j = 0; j < 8; ++j) p.s[j] = xs[j * XSTR + h];
        *(uint4*)&xt[((size_t)(bb * H_DIM + h) << 10) + n0] = p.q;
    }
}

// ---------------------------------------------------------------------------
// Phase-2 helpers: tile loads (A from LDS, 4 B tiles from global) and WMMA.
// ---------------------------------------------------------------------------
__device__ __forceinline__ void load_k(const unsigned short* __restrict__ probs,
                                       const unsigned short* __restrict__ xb,
                                       int k, int m16, int half, int wave,
                                       V16& a, V16* b)
{
    const int abase = m16 * PSTR + k + half * 8;
    a.q[0] = *(const uint4*)&probs[abase];        // ds_load_b128
    a.q[1] = *(const uint4*)&probs[abase + 16];
#pragma unroll
    for (int s = 0; s < 4; ++s) {
        const unsigned short* bp =
            xb + ((size_t)(wave * 64 + s * 16 + m16) << 10) + k + half * 8;
        b[s].q[0] = *(const uint4*)bp;            // global_load_b128 x2
        b[s].q[1] = *(const uint4*)(bp + 16);
    }
}

__device__ __forceinline__ void mm4(const V16& a, const V16* b, v8f* acc)
{
#pragma unroll
    for (int s = 0; s < 4; ++s)
        acc[s] = __builtin_amdgcn_wmma_f32_16x16x32_bf16(
            false, a.v, false, b[s].v, (short)0, acc[s], false, false);
}

// ---------------------------------------------------------------------------
// Kernel 2: one workgroup per (batch b, 16-row i-tile).
// Phase 1: fused score/leaky-relu/mask/softmax-numerator into LDS (bf16).
// Phase 2: double-buffered WMMA GEMM, out[i,h] = (1/rowsum[i]) * sum_j e*X.
// amdgpu_waves_per_eu(1,4): cap the occupancy target at 4 waves/EU so the
// scheduler/allocator keeps both pipeline buffer sets (~112 VGPRs) live.
// sched_barrier(0) fences stop prefetch loads from sinking to their uses.
// ---------------------------------------------------------------------------
__global__ __launch_bounds__(256)
__attribute__((amdgpu_waves_per_eu(1, 4)))
void gat_main(
    const int* __restrict__ adj, const float* __restrict__ s_src,
    const float* __restrict__ s_dst, const float* __restrict__ attn_b,
    const unsigned short* __restrict__ xt, float* __restrict__ out)
{
    __shared__ __align__(16) unsigned short probs[16 * PSTR]; // exp numerators (bf16)
    __shared__ float sdst_l[N_DIM];
    __shared__ float ssrc_l[16];
    __shared__ float redbuf[256];
    __shared__ float rowmax[16];
    __shared__ float rowsum[16];
    __shared__ float bias_l;

    const int tid = threadIdx.x;
    const int i0  = blockIdx.x * 16;
    const int bb  = blockIdx.y;

    for (int c = tid; c < N_DIM; c += 256) sdst_l[c] = s_dst[bb * N_DIM + c];
    if (tid < 16) ssrc_l[tid] = s_src[bb * N_DIM + i0 + tid];
    if (tid == 0) bias_l = attn_b[0];
    __syncthreads();

    // ---- Phase 1: scores -> masked leaky-relu -> rowmax -> exp -> rowsum ----
    const int r  = tid >> 4;          // row within tile (0..15)
    const int cg = tid & 15;          // column group
    const int i  = i0 + r;
    const int* adjr = adj + ((size_t)bb * N_DIM + i) * N_DIM;
    const float srf = ssrc_l[r], bf = bias_l;

    unsigned long long bits = 0ull;   // per-thread mask cache: adjacency read once
    float mx = -INFINITY;
#pragma unroll 4
    for (int k = 0; k < 64; ++k) {
        int c = cg + (k << 4);
        bool m = (adjr[c] != 0) || (c == i);     // adjacency + self loop
        float s = srf + sdst_l[c] + bf;
        s = (s >= 0.f) ? s : NEG_SLOPE * s;      // leaky relu
        if (m) { mx = fmaxf(mx, s); bits |= (1ull << k); }
    }
    redbuf[tid] = mx;
    __syncthreads();
    if (tid < 16) {
        float m = redbuf[tid * 16];
        for (int g = 1; g < 16; ++g) m = fmaxf(m, redbuf[tid * 16 + g]);
        rowmax[tid] = m;
    }
    __syncthreads();

    const float rm = rowmax[r];
    float sum = 0.f;
#pragma unroll 4
    for (int k = 0; k < 64; ++k) {
        int c = cg + (k << 4);
        float s = srf + sdst_l[c] + bf;
        s = (s >= 0.f) ? s : NEG_SLOPE * s;
        float p = 0.f;
        if ((bits >> k) & 1ull) { p = __expf(s - rm); sum += p; }
        probs[r * PSTR + c] = f32_to_bf16(p);    // unnormalized numerator
    }
    redbuf[tid] = sum;
    __syncthreads();
    if (tid < 16) {
        float s = 0.f;
        for (int g = 0; g < 16; ++g) s += redbuf[tid * 16 + g];
        rowsum[tid] = s;
    }
    __syncthreads();

    // ---- Phase 2: software-pipelined WMMA over K (j) in chunks of 32 ----
    const int wave = tid >> 5, lane = tid & 31;
    const int m16  = lane & 15;       // A: row M / B,C: col N
    const int half = lane >> 4;       // lane-group selector per ISA layout
    v8f zero = {};
    v8f acc[4] = {zero, zero, zero, zero};       // 4 h-tiles (64 columns) per wave
    const unsigned short* xb = xt + ((size_t)bb * H_DIM << 10);

    V16 a0, a1; V16 b0[4], b1[4];
    load_k(probs, xb, 0, m16, half, wave, a0, b0);
    for (int kt = 0; kt < 30; kt += 2) {
        load_k(probs, xb, (kt + 1) << 5, m16, half, wave, a1, b1); // prefetch k+1
        SCHED_FENCE();  // keep k+1 loads above the k WMMAs
        mm4(a0, b0, acc);                                          // consume k
        SCHED_FENCE();
        load_k(probs, xb, (kt + 2) << 5, m16, half, wave, a0, b0); // prefetch k+2
        SCHED_FENCE();  // keep k+2 loads above the k+1 WMMAs
        mm4(a1, b1, acc);                                          // consume k+1
        SCHED_FENCE();
    }
    load_k(probs, xb, 31 << 5, m16, half, wave, a1, b1);
    SCHED_FENCE();
    mm4(a0, b0, acc);
    mm4(a1, b1, acc);

    // ---- Epilogue: softmax denominator scaling (v_rcp) + coalesced f32 store ----
    float inv[8];
#pragma unroll
    for (int e = 0; e < 8; ++e)
        inv[e] = __builtin_amdgcn_rcpf(rowsum[e + half * 8]);
#pragma unroll
    for (int s = 0; s < 4; ++s) {
        const int h = wave * 64 + s * 16 + m16;
        float* op = out + ((size_t)bb * N_DIM + i0 + half * 8) * H_DIM + h;
#pragma unroll
        for (int e = 0; e < 8; ++e)
            op[(size_t)e * H_DIM] = acc[s][e] * inv[e];
    }
}

extern "C" void kernel_launch(void* const* d_in, const int* in_sizes, int n_in,
                              void* d_out, int out_size, void* d_ws, size_t ws_size,
                              hipStream_t stream) {
    (void)in_sizes; (void)n_in; (void)out_size; (void)ws_size;
    const float* x      = (const float*)d_in[0];
    const int*   adj    = (const int*)d_in[1];
    const float* w_src  = (const float*)d_in[2];
    const float* w_dst  = (const float*)d_in[3];
    const float* attn_b = (const float*)d_in[4];
    float* out = (float*)d_out;

    // workspace: s_src[8192] f32 | s_dst[8192] f32 | Xt[8][512][1024] bf16 (8 MB)
    char* ws = (char*)d_ws;
    float* s_src = (float*)ws;
    float* s_dst = (float*)(ws + 32 * 1024);
    unsigned short* xt = (unsigned short*)(ws + 64 * 1024);

    gat_prep<<<dim3((B_DIM * N_DIM) / 8), dim3(256), 0, stream>>>(
        x, w_src, w_dst, s_src, s_dst, xt);
    gat_main<<<dim3(N_DIM / 16, B_DIM), dim3(256), 0, stream>>>(
        adj, s_src, s_dst, attn_b, xt, out);
}
